// EuclideanBranch_29394756174242
// MI455X (gfx1250) — compile-verified, matrix operand order checked by reading, not run on hardware
//
#include <hip/hip_runtime.h>
#include <hip/hip_bf16.h>

typedef __bf16 bf16;
typedef __attribute__((ext_vector_type(16))) __bf16 v16bf;
typedef __attribute__((ext_vector_type(8)))  __bf16 v8bf;
typedef __attribute__((ext_vector_type(8)))  float  v8f;

__device__ inline bf16 f2bf(float f) {
  union { float f; unsigned u; } x; x.f = f;
  unsigned r = (x.u + 0x7FFFu + ((x.u >> 16) & 1u)) >> 16;
  unsigned short s = (unsigned short)r;
  bf16 out; __builtin_memcpy(&out, &s, 2); return out;
}
__device__ inline float bf2f(bf16 b) {
  unsigned short s; __builtin_memcpy(&s, &b, 2);
  union { unsigned u; float f; } x; x.u = ((unsigned)s) << 16; return x.f;
}

// ---------------------------------------------------------------------------
// Templated implicit-GEMM conv3d (normal / strided / 2x transpose) with fused
// BN+ReLU+mask+residual epilogue.
//  * 128-thread block = 4 waves = 4 consecutive 16-voxel M-tiles.
//  * Per tap, the packed WMMA B fragments (all K-chunks, 2 N-tiles) are staged
//    cooperatively into LDS once and consumed by all 4 waves via ds_load_b128.
//  * While staging tap t, threads prefetch tap t+1's weights (global_prefetch).
//  * Each wave: 16 voxels x 32 channels, two v_wmma_f32_16x16x32_bf16 per
//    K-chunk sharing one A fragment (b128 global loads).
// ---------------------------------------------------------------------------
template<int KSZ, int STRIDE, int TRANSP, int DOUT, int CIN, int COUT>
__global__ __launch_bounds__(128)
void conv_wmma_t(const bf16* __restrict__ in, const bf16* __restrict__ wpk,
                 const float* __restrict__ bng, const float* __restrict__ bnb,
                 const float* __restrict__ bnm, const float* __restrict__ bnv,
                 const float* __restrict__ mask,
                 const bf16* __restrict__ res, int res_cs,
                 bf16* __restrict__ out, int out_cs, int out_coff)
{
  constexpr int DIN  = TRANSP ? (DOUT >> 1) : (STRIDE == 2 ? DOUT * 2 : DOUT);
  constexpr int PAD  = (KSZ == 3) ? 1 : 0;
  constexpr int LOG2 = (DOUT == 64) ? 6 : (DOUT == 32) ? 5 : (DOUT == 16) ? 4 : 3;
  constexpr int NCH  = CIN / 32;        // K-chunks of 32
  constexpr int NTT  = COUT / 16;       // N-tiles in packed weights
  constexpr int NTAP = KSZ * KSZ * KSZ;
  constexpr int NWAVE = 4;

  // per-tap staged B fragments: NCH chunks x (2 tiles x 32 lanes x 16 bf16)
  __shared__ __align__(32) bf16 bsm[NCH * 1024];

  const int tid    = threadIdx.x;
  const int lane   = tid & 31;
  const int wv     = tid >> 5;
  const int col    = lane & 15;
  const int khalf8 = (lane & 16) ? 8 : 0;         // ISA 16-bit A/B K-half
  const int mbase  = (blockIdx.x * NWAVE + wv) << 4;
  const int mv     = mbase + col;                 // Vout % 64 == 0 always
  const int oz     = mv & (DOUT - 1);
  const int oy     = (mv >> LOG2) & (DOUT - 1);
  const int ox     = mv >> (2 * LOG2);
  const int n0     = (blockIdx.y << 5) + col;

  v8f acc0 = {}, acc1 = {};

  #pragma unroll 1
  for (int t = 0; t < NTAP; ++t) {
    // ---- cooperative B stage into LDS (+ prefetch of next tap) ----
    __syncthreads();   // previous-tap consumers done
    {
      // per kcc: contiguous 1024 elems (2 N-tiles) at ((t*NCH+kcc)*NTT + 2*by)*512
      #pragma unroll
      for (int c = tid; c < NCH * 128; c += 128) {
        const int kcc = c >> 7;
        const int offs = (c & 127) * 8;
        const bf16* src = wpk + ((size_t)((t * NCH + kcc) * NTT + (blockIdx.y << 1))) * 512 + offs;
        *(v8bf*)(bsm + kcc * 1024 + offs) = *(const v8bf*)src;
        if (t + 1 < NTAP) __builtin_prefetch(src + (size_t)NCH * NTT * 512, 0, 1);
      }
    }
    __syncthreads();   // B fragments visible to all waves

    // ---- per-wave A fragment address for this tap ----
    const int kc2 = t % KSZ, tt = t / KSZ, kb = tt % KSZ, ka = tt / KSZ;
    bool v; int ix, iy, iz;
    if (TRANSP) {
      v  = ((ox & 1) == ka) && ((oy & 1) == kb) && ((oz & 1) == kc2);
      ix = (ox - ka) >> 1; iy = (oy - kb) >> 1; iz = (oz - kc2) >> 1;
    } else {
      ix = ox * STRIDE + ka - PAD; iy = oy * STRIDE + kb - PAD; iz = oz * STRIDE + kc2 - PAD;
      v  = (unsigned)ix < (unsigned)DIN && (unsigned)iy < (unsigned)DIN && (unsigned)iz < (unsigned)DIN;
    }
    const int vlin = v ? ((ix * DIN + iy) * DIN + iz) : 0;
    const bf16* ap = in + (size_t)vlin * CIN + khalf8;

    #pragma unroll
    for (int kcc = 0; kcc < NCH; ++kcc) {
      v16bf a = {};
      if (v) {
        const v8bf* p = (const v8bf*)(ap + kcc * 32);
        v8bf lo = p[0];      // channels kc+khalf    .. +7   (16B)
        v8bf hi = p[2];      // channels kc+khalf+16 .. +23  (16B)
        a = __builtin_shufflevector(lo, hi, 0,1,2,3,4,5,6,7,8,9,10,11,12,13,14,15);
      }
      const v16bf* bp = (const v16bf*)(bsm + kcc * 1024);
      v16bf b0 = bp[lane];        // N-tile 0 fragment (ds_load_b128 x2)
      v16bf b1 = bp[32 + lane];   // N-tile 1 fragment
      acc0 = __builtin_amdgcn_wmma_f32_16x16x32_bf16(false, a, false, b0, (short)0, acc0, false, false);
      acc1 = __builtin_amdgcn_wmma_f32_16x16x32_bf16(false, a, false, b1, (short)0, acc1, false, false);
    }
  }

  // ---- epilogue: BN + ReLU + mask + residual (C/D layout: lane=col, 8 rows)
  const int n1 = n0 + 16;
  float s0 = 1.f, o0 = 0.f, s1 = 1.f, o1 = 0.f;
  if (bng) {
    s0 = bng[n0] * rsqrtf(bnv[n0] + 1e-5f); o0 = bnb[n0] - bnm[n0] * s0;
    s1 = bng[n1] * rsqrtf(bnv[n1] + 1e-5f); o1 = bnb[n1] - bnm[n1] * s1;
  }
  const int rbase = (lane & 16) ? 8 : 0;
  #pragma unroll
  for (int r = 0; r < 8; ++r) {
    const int vox = mbase + rbase + r;
    float x0 = acc0[r], x1 = acc1[r];
    if (bng) { x0 = fmaxf(x0 * s0 + o0, 0.f); x1 = fmaxf(x1 * s1 + o1, 0.f); }
    if (mask) { float mm = mask[vox]; x0 *= mm; x1 *= mm; }
    if (res)  { x0 += bf2f(res[(size_t)vox * res_cs + n0]);
                x1 += bf2f(res[(size_t)vox * res_cs + n1]); }
    bf16* o = out + (size_t)vox * out_cs + out_coff;
    o[n0] = f2bf(x0); o[n1] = f2bf(x1);
  }
}

template<int KSZ, int STRIDE, int TRANSP, int DOUT, int CIN, int COUT>
static void launch_conv(const bf16* in, const bf16* wpk,
                        const float* g, const float* b, const float* m, const float* v,
                        const float* mask, const bf16* res, int res_cs,
                        bf16* out, int out_cs, int out_coff, hipStream_t stream) {
  dim3 grid((unsigned)(DOUT * DOUT * DOUT / 64), (unsigned)(COUT / 32));
  conv_wmma_t<KSZ, STRIDE, TRANSP, DOUT, CIN, COUT><<<grid, 128, 0, stream>>>(
      in, wpk, g, b, m, v, mask, res, res_cs, out, out_cs, out_coff);
}

// ---------------------------------------------------------------------------
// Support kernels
// ---------------------------------------------------------------------------
__global__ void zero_f32_kernel(float* p, int n) {
  int i = blockIdx.x * blockDim.x + threadIdx.x;
  if (i < n) p[i] = 0.f;
}

__global__ void scatter_kernel(const float* __restrict__ feats, const int* __restrict__ coords,
                               float* __restrict__ gf, float* __restrict__ m1, int N) {
  int p = blockIdx.x * blockDim.x + threadIdx.x;
  if (p >= N) return;
  int cx = coords[3 * p + 0], cy = coords[3 * p + 1], cz = coords[3 * p + 2];
  int v = (cx * 64 + cy) * 64 + cz;
  atomicAdd(&gf[v * 3 + 0], feats[3 * p + 0]);
  atomicAdd(&gf[v * 3 + 1], feats[3 * p + 1]);
  atomicAdd(&gf[v * 3 + 2], feats[3 * p + 2]);
  m1[v] = 1.0f;
}

__global__ void pad_grid_kernel(const float* __restrict__ gf, bf16* __restrict__ gin, int V) {
  int i = blockIdx.x * blockDim.x + threadIdx.x;
  if (i >= V * 32) return;
  int v = i >> 5, c = i & 31;
  gin[i] = (c < 3) ? f2bf(gf[v * 3 + c]) : f2bf(0.f);
}

__global__ void pool_mask_kernel(const float* __restrict__ src, float* __restrict__ dst, int Do) {
  int i = blockIdx.x * blockDim.x + threadIdx.x;
  int Vo = Do * Do * Do; if (i >= Vo) return;
  int c = i % Do; int t = i / Do; int b = t % Do; int a = t / Do;
  int Di = Do * 2; float m = 0.f;
  for (int da = 0; da < 2; ++da)
    for (int db = 0; db < 2; ++db)
      for (int dc = 0; dc < 2; ++dc)
        m = fmaxf(m, src[((a * 2 + da) * Di + (b * 2 + db)) * Di + (c * 2 + dc)]);
  dst[i] = m;
}

// f32 weights (taps, CinReal, Cout) -> bf16 packed per-lane WMMA B fragments:
// index = (((t*NCH + kcc)*NTT + ntile)*32 + lane)*16 + e
__global__ void wpack_kernel(const float* __restrict__ src, bf16* __restrict__ dst,
                             int taps, int cinreal, int cinpad, int cout) {
  int i = blockIdx.x * blockDim.x + threadIdx.x;
  int ntt = cout >> 4, nch = cinpad >> 5;
  int total = taps * nch * ntt * 32 * 16;
  if (i >= total) return;
  int e = i & 15;
  int l = (i >> 4) & 31;
  int rest = i >> 9;
  int nt = rest % ntt; rest /= ntt;
  int kcc = rest % nch;
  int t = rest / nch;
  int khalf = (l & 16) ? 8 : 0;
  int k = kcc * 32 + khalf + (e < 8 ? e : e + 8);   // e>=8 -> +16 offset half
  int n = nt * 16 + (l & 15);
  float v = (k < cinreal) ? src[(t * cinreal + k) * cout + n] : 0.f;
  dst[i] = f2bf(v);
}

// channel-strided bf16 copy (for skip concats)
__global__ void copy_ch_kernel(const bf16* __restrict__ src, int scs,
                               bf16* __restrict__ dst, int dcs, int dcoff, int C, int V) {
  int i = blockIdx.x * blockDim.x + threadIdx.x;
  if (i >= V * C) return;
  int v = i / C, c = i % C;
  dst[(size_t)v * dcs + dcoff + c] = src[(size_t)v * scs + c];
}

// final 1x1 conv (32 -> 20) + bias + gather at point coords, f32 out
__global__ void out_gather_kernel(const bf16* __restrict__ d1, const int* __restrict__ coords,
                                  const float* __restrict__ ow, const float* __restrict__ ob,
                                  float* __restrict__ out, int N) {
  int p = blockIdx.x * blockDim.x + threadIdx.x;
  if (p >= N) return;
  int cx = coords[3 * p], cy = coords[3 * p + 1], cz = coords[3 * p + 2];
  const bf16* f = d1 + (size_t)((cx * 64 + cy) * 64 + cz) * 32;
  float fv[32];
  #pragma unroll
  for (int k = 0; k < 32; ++k) fv[k] = bf2f(f[k]);
  for (int c = 0; c < 20; ++c) {
    float s = ob[c];
    #pragma unroll
    for (int k = 0; k < 32; ++k) s += fv[k] * ow[k * 20 + c];
    out[(size_t)p * 20 + c] = s;
  }
}

// ---------------------------------------------------------------------------
// Host orchestration
// ---------------------------------------------------------------------------
extern "C" void kernel_launch(void* const* d_in, const int* in_sizes, int n_in,
                              void* d_out, int out_size, void* d_ws, size_t ws_size,
                              hipStream_t stream) {
  (void)n_in; (void)out_size; (void)ws_size;
  const int N = in_sizes[0] / 3;
  const int D1 = 64, D2 = 32, D3 = 16, D4 = 8;
  const int V1 = D1 * D1 * D1, V2 = D2 * D2 * D2, V3 = D3 * D3 * D3, V4 = D4 * D4 * D4;

  const float* feats  = (const float*)d_in[0];
  const int*   coords = (const int*)d_in[1];

  // ---- workspace bump allocator ----
  char* base = (char*)d_ws; size_t off = 0;
  auto alloc = [&](size_t bytes) -> void* {
    void* p = base + off; off = (off + bytes + 255) & ~(size_t)255; return p;
  };
  float* gf = (float*)alloc((size_t)V1 * 3 * 4);
  float* m1 = (float*)alloc((size_t)V1 * 4);
  float* m2 = (float*)alloc((size_t)V2 * 4);
  float* m3 = (float*)alloc((size_t)V3 * 4);
  float* m4 = (float*)alloc((size_t)V4 * 4);
  auto abuf = [&](int V, int C) { return (bf16*)alloc((size_t)V * C * 2); };
  bf16* gin = abuf(V1, 32);
  bf16* x1  = abuf(V1, 32); bf16* t1 = abuf(V1, 32); bf16* e1 = abuf(V1, 32);
  bf16* x2  = abuf(V2, 64); bf16* t2 = abuf(V2, 64); bf16* e2 = abuf(V2, 64);
  bf16* x3  = abuf(V3, 96); bf16* t3 = abuf(V3, 96); bf16* e3 = abuf(V3, 96);
  bf16* xm  = abuf(V4, 128); bf16* tm = abuf(V4, 128); bf16* xmid = abuf(V4, 128);
  bf16* y3  = abuf(V3, 192); bf16* r3 = abuf(V3, 96); bf16* d3t = abuf(V3, 96); bf16* d3 = abuf(V3, 96);
  bf16* y2  = abuf(V2, 128); bf16* r2 = abuf(V2, 64); bf16* d2t = abuf(V2, 64); bf16* d2 = abuf(V2, 64);
  bf16* y1  = abuf(V1, 64);  bf16* r1 = abuf(V1, 32); bf16* d1t = abuf(V1, 32); bf16* d1 = abuf(V1, 32);

  // ---- weight table: {d_in idx, taps, cin_real, cin_pad, cout} ----
  struct WL { int idx, taps, cin, cinpad, cout; };
  enum { L_IN, L_E1A, L_E1B, L_DN1, L_E2A, L_E2B, L_DN2, L_E3A, L_E3B, L_DN3,
         L_MA, L_MB, L_UP3, L_LIN3, L_D3A, L_D3B, L_UP2, L_LIN2, L_D2A, L_D2B,
         L_UP1, L_LIN1, L_D1A, L_D1B, NLAYER };
  const WL wl[NLAYER] = {
    {  2, 27,   3,  32,  32}, {  7, 27,  32,  32,  32}, { 12, 27,  32,  32,  32},
    { 17,  8,  32,  32,  64}, { 22, 27,  64,  64,  64}, { 27, 27,  64,  64,  64},
    { 32,  8,  64,  64,  96}, { 37, 27,  96,  96,  96}, { 42, 27,  96,  96,  96},
    { 47,  8,  96,  96, 128}, { 52, 27, 128, 128, 128}, { 57, 27, 128, 128, 128},
    { 62,  8, 128, 128,  96}, { 67,  1, 192, 192,  96}, { 68, 27, 192, 192,  96},
    { 73, 27,  96,  96,  96}, { 78,  8,  96,  96,  64}, { 83,  1, 128, 128,  64},
    { 84, 27, 128, 128,  64}, { 89, 27,  64,  64,  64}, { 94,  8,  64,  64,  32},
    { 99,  1,  64,  64,  32}, {100, 27,  64,  64,  32}, {105, 27,  32,  32,  32},
  };
  bf16* wbf[NLAYER];
  for (int l = 0; l < NLAYER; ++l) {
    int tot = wl[l].taps * wl[l].cinpad * wl[l].cout;
    wbf[l] = (bf16*)alloc((size_t)tot * 2);
    wpack_kernel<<<(tot + 255) / 256, 256, 0, stream>>>(
        (const float*)d_in[wl[l].idx], wbf[l], wl[l].taps, wl[l].cin, wl[l].cinpad, wl[l].cout);
  }
  auto G = [&](int l) { return (const float*)d_in[wl[l].idx + 1]; };
  auto B = [&](int l) { return (const float*)d_in[wl[l].idx + 2]; };
  auto M = [&](int l) { return (const float*)d_in[wl[l].idx + 3]; };
  auto V = [&](int l) { return (const float*)d_in[wl[l].idx + 4]; };

  // ---- build dense grid + masks ----
  zero_f32_kernel<<<(V1 * 3 + 255) / 256, 256, 0, stream>>>(gf, V1 * 3);
  zero_f32_kernel<<<(V1 + 255) / 256, 256, 0, stream>>>(m1, V1);
  scatter_kernel<<<(N + 255) / 256, 256, 0, stream>>>(feats, coords, gf, m1, N);
  pad_grid_kernel<<<(V1 * 32 + 255) / 256, 256, 0, stream>>>(gf, gin, V1);
  pool_mask_kernel<<<(V2 + 255) / 256, 256, 0, stream>>>(m1, m2, D2);
  pool_mask_kernel<<<(V3 + 255) / 256, 256, 0, stream>>>(m2, m3, D3);
  pool_mask_kernel<<<(V4 + 255) / 256, 256, 0, stream>>>(m3, m4, D4);

  auto ccopy = [&](const bf16* src, int scs, bf16* dst, int dcs, int dcoff, int C, int Vv) {
    copy_ch_kernel<<<(Vv * C + 255) / 256, 256, 0, stream>>>(src, scs, dst, dcs, dcoff, C, Vv);
  };

  // ---- encoder ----
  launch_conv<3,1,0,64, 32, 32>(gin, wbf[L_IN ], G(L_IN ),B(L_IN ),M(L_IN ),V(L_IN ), m1, nullptr,0,  x1, 32,0, stream);
  launch_conv<3,1,0,64, 32, 32>(x1,  wbf[L_E1A], G(L_E1A),B(L_E1A),M(L_E1A),V(L_E1A), m1, nullptr,0,  t1, 32,0, stream);
  launch_conv<3,1,0,64, 32, 32>(t1,  wbf[L_E1B], G(L_E1B),B(L_E1B),M(L_E1B),V(L_E1B), m1, x1,32,      e1, 32,0, stream);
  launch_conv<2,2,0,32, 32, 64>(e1,  wbf[L_DN1], G(L_DN1),B(L_DN1),M(L_DN1),V(L_DN1), m2, nullptr,0,  x2, 64,0, stream);
  launch_conv<3,1,0,32, 64, 64>(x2,  wbf[L_E2A], G(L_E2A),B(L_E2A),M(L_E2A),V(L_E2A), m2, nullptr,0,  t2, 64,0, stream);
  launch_conv<3,1,0,32, 64, 64>(t2,  wbf[L_E2B], G(L_E2B),B(L_E2B),M(L_E2B),V(L_E2B), m2, x2,64,      e2, 64,0, stream);
  launch_conv<2,2,0,16, 64, 96>(e2,  wbf[L_DN2], G(L_DN2),B(L_DN2),M(L_DN2),V(L_DN2), m3, nullptr,0,  x3, 96,0, stream);
  launch_conv<3,1,0,16, 96, 96>(x3,  wbf[L_E3A], G(L_E3A),B(L_E3A),M(L_E3A),V(L_E3A), m3, nullptr,0,  t3, 96,0, stream);
  launch_conv<3,1,0,16, 96, 96>(t3,  wbf[L_E3B], G(L_E3B),B(L_E3B),M(L_E3B),V(L_E3B), m3, x3,96,      e3, 96,0, stream);
  launch_conv<2,2,0, 8, 96,128>(e3,  wbf[L_DN3], G(L_DN3),B(L_DN3),M(L_DN3),V(L_DN3), m4, nullptr,0,  xm,128,0, stream);
  launch_conv<3,1,0, 8,128,128>(xm,  wbf[L_MA ], G(L_MA ),B(L_MA ),M(L_MA ),V(L_MA ), m4, nullptr,0,  tm,128,0, stream);
  launch_conv<3,1,0, 8,128,128>(tm,  wbf[L_MB ], G(L_MB ),B(L_MB ),M(L_MB ),V(L_MB ), m4, xm,128,     xmid,128,0, stream);

  // ---- decoder level 3 ----
  launch_conv<2,2,1,16,128, 96>(xmid,wbf[L_UP3], G(L_UP3),B(L_UP3),M(L_UP3),V(L_UP3), m3, nullptr,0,  y3,192,0, stream);
  ccopy(e3, 96, y3, 192, 96, 96, V3);
  launch_conv<1,1,0,16,192, 96>(y3,  wbf[L_LIN3], nullptr,nullptr,nullptr,nullptr, nullptr, nullptr,0, r3, 96,0, stream);
  launch_conv<3,1,0,16,192, 96>(y3,  wbf[L_D3A], G(L_D3A),B(L_D3A),M(L_D3A),V(L_D3A), m3, nullptr,0,  d3t,96,0, stream);
  launch_conv<3,1,0,16, 96, 96>(d3t, wbf[L_D3B], G(L_D3B),B(L_D3B),M(L_D3B),V(L_D3B), m3, r3,96,      d3, 96,0, stream);

  // ---- decoder level 2 ----
  launch_conv<2,2,1,32, 96, 64>(d3,  wbf[L_UP2], G(L_UP2),B(L_UP2),M(L_UP2),V(L_UP2), m2, nullptr,0,  y2,128,0, stream);
  ccopy(e2, 64, y2, 128, 64, 64, V2);
  launch_conv<1,1,0,32,128, 64>(y2,  wbf[L_LIN2], nullptr,nullptr,nullptr,nullptr, nullptr, nullptr,0, r2, 64,0, stream);
  launch_conv<3,1,0,32,128, 64>(y2,  wbf[L_D2A], G(L_D2A),B(L_D2A),M(L_D2A),V(L_D2A), m2, nullptr,0,  d2t,64,0, stream);
  launch_conv<3,1,0,32, 64, 64>(d2t, wbf[L_D2B], G(L_D2B),B(L_D2B),M(L_D2B),V(L_D2B), m2, r2,64,      d2, 64,0, stream);

  // ---- decoder level 1 ----
  launch_conv<2,2,1,64, 64, 32>(d2,  wbf[L_UP1], G(L_UP1),B(L_UP1),M(L_UP1),V(L_UP1), m1, nullptr,0,  y1, 64,0, stream);
  ccopy(e1, 32, y1, 64, 32, 32, V1);
  launch_conv<1,1,0,64, 64, 32>(y1,  wbf[L_LIN1], nullptr,nullptr,nullptr,nullptr, nullptr, nullptr,0, r1, 32,0, stream);
  launch_conv<3,1,0,64, 64, 32>(y1,  wbf[L_D1A], G(L_D1A),B(L_D1A),M(L_D1A),V(L_D1A), m1, nullptr,0,  d1t,32,0, stream);
  launch_conv<3,1,0,64, 32, 32>(d1t, wbf[L_D1B], G(L_D1B),B(L_D1B),M(L_D1B),V(L_D1B), m1, r1,32,      d1, 32,0, stream);

  // ---- output head ----
  out_gather_kernel<<<(N + 127) / 128, 128, 0, stream>>>(
      d1, coords, (const float*)d_in[110], (const float*)d_in[111], (float*)d_out, N);
}
